// HybridGPSBlock_66949950210411
// MI455X (gfx1250) — matrix operand order repaired and checked by reading, not compiled, not run on hardware
//
#include <hip/hip_runtime.h>
#include <hip/hip_bf16.h>

typedef unsigned int u32;
typedef __attribute__((ext_vector_type(16))) __bf16 v16bf;
typedef __attribute__((ext_vector_type(8)))  float  v8f;

#define Bx 4
#define Lx 2048
#define Dx 256
#define Hx 8
#define Mx (Bx * Lx)   // 8192 tokens

// ---------------- helpers ----------------
__device__ inline float gelu_f(float x) {
    return 0.5f * x * (1.0f + erff(x * 0.70710678118654752f));
}
__device__ inline u32 f2bf_rne(float f) {          // exact RNE (weight prep only)
    u32 u = __float_as_uint(f);
    return (u + 0x7FFFu + ((u >> 16) & 1u)) >> 16;
}
__device__ inline u32 packbf_rne(float a, float b) {
    return f2bf_rne(a) | (f2bf_rne(b) << 16);
}
__device__ inline u32 packbf_fast(float a, float b) {   // round-half-up, 3 ops
    u32 ua = __float_as_uint(a), ub = __float_as_uint(b);
    return ((ua + 0x8000u) >> 16) | ((ub + 0x8000u) & 0xFFFF0000u);
}
__device__ inline float block_reduce_sum(float v, float* red) {
    int t = threadIdx.x;
    red[t] = v; __syncthreads();
    for (int o = 128; o > 0; o >>= 1) { if (t < o) red[t] += red[t + o]; __syncthreads(); }
    float r = red[0]; __syncthreads();
    return r;
}

// ---------------- LayerNorm (one 256-thread block per row, D=256) ----------------
__global__ __launch_bounds__(256) void ln_kernel(const float* __restrict__ x,
                                                 const float* __restrict__ g,
                                                 const float* __restrict__ b,
                                                 float* __restrict__ out) {
    __shared__ float red[256];
    int row = blockIdx.x, t = threadIdx.x;
    size_t i = (size_t)row * Dx + t;
    float v = x[i];
    float mu = block_reduce_sum(v, red) * (1.0f / Dx);
    float dv = v - mu;
    float var = block_reduce_sum(dv * dv, red) * (1.0f / Dx);
    out[i] = dv * rsqrtf(var + 1e-5f) * g[t] + b[t];
}

// ---------------- Weight prep: W[K][N] f32 (row stride S) -> Wt[n][K/2] packed bf16 pairs ----------------
__global__ void prep_w(const float* __restrict__ src, int srcStride, int K, int N,
                       u32* __restrict__ dst, int kpitch) {
    int i = blockIdx.x * blockDim.x + threadIdx.x;
    int kh = K >> 1;
    if (i >= N * kh) return;
    int n = i / kh, kk = i - n * kh;
    float a = src[(size_t)(2 * kk) * srcStride + n];
    float b = src[(size_t)(2 * kk + 1) * srcStride + n];
    dst[(size_t)n * kpitch + kk] = packbf_rne(a, b);
}

// ---------------- concat bias vector for the big GEMM ----------------
__global__ void biascat_kernel(const float* __restrict__ bf1, const float* __restrict__ bg1,
                               float* __restrict__ bias) {
    int c = blockIdx.x * blockDim.x + threadIdx.x;
    if (c >= 2048) return;
    float v = 0.0f;
    if (c >= 1536 && c < 1792) v = bf1[c - 1536];
    else if (c >= 1792)        v = bg1[c - 1792];
    bias[c] = v;
}

// ---------------- rel-pos bias table (65 entries) ----------------
__global__ __launch_bounds__(256) void relbias_kernel(const float* __restrict__ emb,
                                                      const float* __restrict__ we1,
                                                      const float* __restrict__ be1,
                                                      const float* __restrict__ we2,
                                                      const float* __restrict__ be2,
                                                      float* __restrict__ table) {
    __shared__ float red[256];
    int r = blockIdx.x, j = threadIdx.x;
    float s = be1[j];
    for (int i = 0; i < 64; ++i) s += emb[r * 64 + i] * we1[i * 256 + j];
    float tot = block_reduce_sum(gelu_f(s) * we2[j], red);
    if (j == 0) table[r] = tot + be2[0];
}

// ---------------- WMMA bf16 GEMM ----------------
// C[M,N](ldc) = act( A[M,K](lda, f32) @ Wt + bias [+Cold] )
// Block 256 thr = 8 waves; tile BM=128 BN=128 BK=32; wave = 32x64 out (2x4 WMMA tiles)
// flags: bit0 = accumulate into existing C, act = (flags>>1)&3 : 0 none, 1 gelu, 2 sigmoid
__global__ __launch_bounds__(256) void gemm_bf16(const float* __restrict__ A, int lda,
                                                 const u32* __restrict__ Wt, int kpitch,
                                                 const float* __restrict__ bias,
                                                 float* __restrict__ C, int ldc,
                                                 int M, int N, int K, int flags) {
    __shared__ u32 lA[128][17];   // 128 rows x 16 packed-bf16 pairs (K=32), +1 pad
    __shared__ u32 lB[128][17];   // 128 cols x 16 packed pairs

    int t = threadIdx.x;
    int lane = t & 31, wave = t >> 5;
    int wm = (wave & 3) * 32;          // 4 waves along M
    int wn = (wave >> 2) * 64;         // 2 waves along N
    int rowBase = blockIdx.y * 128;
    int colBase = blockIdx.x * 128;

    v8f acc[2][4] = {};

    int aRow = t >> 1;                 // 0..127
    int aOff = (t & 1) * 16;           // float offset within K-tile: 0 or 16
    int bN   = t >> 1;                 // 0..127
    int bKK  = (t & 1) * 8;            // u32 offset: 0 or 8

    for (int k0 = 0; k0 < K; k0 += 32) {
        // ---- stage A tile (f32 -> packed bf16), 16 floats per thread ----
        {
            int r = rowBase + aRow;
            int c0 = aOff >> 1;
            if (r < M) {
                const float4* ap = reinterpret_cast<const float4*>(A + (size_t)r * lda + k0 + aOff);
                float4 a0 = ap[0], a1 = ap[1], a2 = ap[2], a3 = ap[3];
                lA[aRow][c0 + 0] = packbf_fast(a0.x, a0.y);
                lA[aRow][c0 + 1] = packbf_fast(a0.z, a0.w);
                lA[aRow][c0 + 2] = packbf_fast(a1.x, a1.y);
                lA[aRow][c0 + 3] = packbf_fast(a1.z, a1.w);
                lA[aRow][c0 + 4] = packbf_fast(a2.x, a2.y);
                lA[aRow][c0 + 5] = packbf_fast(a2.z, a2.w);
                lA[aRow][c0 + 6] = packbf_fast(a3.x, a3.y);
                lA[aRow][c0 + 7] = packbf_fast(a3.z, a3.w);
                if (k0 + 32 < K)   // speculative prefetch of next K-tile
                    __builtin_prefetch(A + (size_t)r * lda + k0 + 32 + aOff, 0, 0);
            } else {
                #pragma unroll
                for (int j = 0; j < 8; ++j) lA[aRow][c0 + j] = 0u;
            }
        }
        // ---- stage B tile (already packed bf16 pairs, [n][K/2]), 8 u32 per thread ----
        {
            int col = colBase + bN;
            uint4 w0 = {0, 0, 0, 0}, w1 = {0, 0, 0, 0};
            if (col < N) {
                const u32* wp = Wt + (size_t)col * kpitch + (k0 >> 1) + bKK;
                w0 = reinterpret_cast<const uint4*>(wp)[0];
                w1 = reinterpret_cast<const uint4*>(wp)[1];
            }
            lB[bN][bKK + 0] = w0.x; lB[bN][bKK + 1] = w0.y;
            lB[bN][bKK + 2] = w0.z; lB[bN][bKK + 3] = w0.w;
            lB[bN][bKK + 4] = w1.x; lB[bN][bKK + 5] = w1.y;
            lB[bN][bKK + 6] = w1.z; lB[bN][bKK + 7] = w1.w;
        }
        __syncthreads();

        // ---- fragments per ISA 7.12.2 (16-bit A 16x32, B 32x16) ----
        int r = lane & 15, kh = lane >> 4;
        union Frag { v16bf v; u32 u[8]; };
        Frag fa[2], fb[4];
        #pragma unroll
        for (int j = 0; j < 8; ++j) {
            int kkA = (j & 3) + kh * 4 + (j >> 2) * 8;   // A: lanes0-15 K0..7,16..23 ; lanes16-31 K8..15,24..31
            int kkB = (j & 3) + kh * 4 + (j >> 2) * 8;   // B: lanes0-15 K0..7,16..23 ; lanes16-31 K8..15,24..31
            fa[0].u[j] = lA[wm +      r][kkA];
            fa[1].u[j] = lA[wm + 16 + r][kkA];
            fb[0].u[j] = lB[wn +      r][kkB];
            fb[1].u[j] = lB[wn + 16 + r][kkB];
            fb[2].u[j] = lB[wn + 32 + r][kkB];
            fb[3].u[j] = lB[wn + 48 + r][kkB];
        }
        #pragma unroll
        for (int ti = 0; ti < 2; ++ti) {
            #pragma unroll
            for (int tj = 0; tj < 4; ++tj) {
                acc[ti][tj] = __builtin_amdgcn_wmma_f32_16x16x32_bf16(
                    false, fa[ti].v, false, fb[tj].v, (short)0, acc[ti][tj], false, false);
            }
        }
        __syncthreads();
    }

    // ---- epilogue: tile(ti,tj): row = base+wm+ti*16+(lane>>4)*8+j, col = base+wn+tj*16+(lane&15) ----
    int act = (flags >> 1) & 3;
    int colL = lane & 15, rowL = (lane >> 4) * 8;
    #pragma unroll
    for (int ti = 0; ti < 2; ++ti) {
        #pragma unroll
        for (int tj = 0; tj < 4; ++tj) {
            int col = colBase + wn + tj * 16 + colL;
            if (col >= N) continue;
            float bv = bias ? bias[col] : 0.0f;
            int rb = rowBase + wm + ti * 16 + rowL;
            #pragma unroll
            for (int j = 0; j < 8; ++j) {
                int row = rb + j;
                if (row >= M) continue;
                float v = acc[ti][tj][j];
                if (flags & 1) v += C[(size_t)row * ldc + col];
                v += bv;
                if (act == 1)      v = gelu_f(v);
                else if (act == 2) v = 1.0f / (1.0f + expf(-v));
                C[(size_t)row * ldc + col] = v;
            }
        }
    }
}

// ---------------- neighbor attention: one wave per token, lane = neighbor (K=32=wave32) ----------------
__global__ __launch_bounds__(256) void attn_kernel(const float* __restrict__ G1,
                                                   const int* __restrict__ nbr_idx,
                                                   const int* __restrict__ nbr_mask,
                                                   const int* __restrict__ mask,
                                                   const int* __restrict__ rel_pos,
                                                   const float* __restrict__ table,
                                                   float* __restrict__ agg0) {
    int wave = threadIdx.x >> 5, lane = threadIdx.x & 31;
    int token = blockIdx.x * 8 + wave;
    int b = token / Lx;
    int rowBase = b * Lx;

    int idx = nbr_idx[(size_t)token * 32 + lane];
    int nm  = nbr_mask[(size_t)token * 32 + lane] * mask[token];
    int rp  = rel_pos[(size_t)token * 32 + lane];
    rp = min(max(rp, 0), 64);

    const float* qrow = G1 + (size_t)token * 2048;                   // q at col 0
    const float* krow = G1 + (size_t)(rowBase + idx) * 2048 + 256;   // hk at col 256
    float dot = 0.0f;
    for (int d = 0; d < 256; d += 4) {
        float4 qv = *reinterpret_cast<const float4*>(qrow + d);
        float4 kv = *reinterpret_cast<const float4*>(krow + d);
        dot += qv.x*kv.x + qv.y*kv.y + qv.z*kv.z + qv.w*kv.w;
    }
    float logit = dot * 0.0625f + table[rp];   // 1/sqrt(256)
    if (!nm) logit = -1e30f;

    float mx = logit;
    for (int o = 16; o > 0; o >>= 1) mx = fmaxf(mx, __shfl_xor(mx, o));
    float e = expf(logit - mx);
    float se = e;
    for (int o = 16; o > 0; o >>= 1) se += __shfl_xor(se, o);
    float w = e / se;

    float av[8];
    #pragma unroll
    for (int j = 0; j < 8; ++j) av[j] = 0.0f;
    for (int k = 0; k < 32; ++k) {
        float wk = __shfl(w, k);
        int   ik = __shfl(idx, k);
        const float* vr = G1 + (size_t)(rowBase + ik) * 2048 + 512 + lane * 8;  // hv at col 512
        float4 a = *reinterpret_cast<const float4*>(vr);
        float4 c = *reinterpret_cast<const float4*>(vr + 4);
        av[0] += wk*a.x; av[1] += wk*a.y; av[2] += wk*a.z; av[3] += wk*a.w;
        av[4] += wk*c.x; av[5] += wk*c.y; av[6] += wk*c.z; av[7] += wk*c.w;
    }
    float* o = agg0 + (size_t)token * 256 + lane * 8;
    *reinterpret_cast<float4*>(o)     = make_float4(av[0], av[1], av[2], av[3]);
    *reinterpret_cast<float4*>(o + 4) = make_float4(av[4], av[5], av[6], av[7]);
}

// ---------------- h_local = (h + g*agg) * mf ----------------
__global__ void hlocal_kernel(const float* __restrict__ h, const float* __restrict__ g,
                              const float* __restrict__ agg, const int* __restrict__ mask,
                              float* __restrict__ hl) {
    size_t i = (size_t)blockIdx.x * blockDim.x + threadIdx.x;
    float mf = (float)mask[i >> 8];
    hl[i] = (h[i] + g[i] * agg[i]) * mf;
}

// ---------------- linear attention: kv[b,h][32][32] + ksum[b,h][32] ----------------
__global__ __launch_bounds__(256) void kv_kernel(const float* __restrict__ G1,
                                                 const int* __restrict__ mask,
                                                 float* __restrict__ kv,
                                                 float* __restrict__ ksum) {
    __shared__ float kgs[32][33], vgs[32][33];
    int bh = blockIdx.x, b = bh >> 3, h = bh & 7;
    int t = threadIdx.x;
    int d = t >> 3, mb = (t & 7) * 4;
    float a0 = 0, a1 = 0, a2 = 0, a3 = 0, ks = 0;

    for (int l0 = 0; l0 < Lx; l0 += 32) {
        #pragma unroll
        for (int i = 0; i < 4; ++i) {
            int lin = t + i * 256;
            int ll = lin >> 5, dd = lin & 31;
            int tok = b * Lx + l0 + ll;
            float mf = (float)mask[tok];
            float kr = G1[(size_t)tok * 2048 + 1024 + h * 32 + dd];  // kg' at col 1024
            kgs[ll][dd] = (kr > 0.0f ? kr + 1.0f : expf(kr)) * mf;   // phi = elu+1
            vgs[ll][dd] = G1[(size_t)tok * 2048 + 1280 + h * 32 + dd] * mf; // vg' at 1280
        }
        __syncthreads();
        #pragma unroll 4
        for (int ll = 0; ll < 32; ++ll) {
            float kd = kgs[ll][d];
            ks += kd;
            a0 += kd * vgs[ll][mb + 0];
            a1 += kd * vgs[ll][mb + 1];
            a2 += kd * vgs[ll][mb + 2];
            a3 += kd * vgs[ll][mb + 3];
        }
        __syncthreads();
    }
    float* kvp = kv + ((size_t)bh * 32 + d) * 32 + mb;
    kvp[0] = a0; kvp[1] = a1; kvp[2] = a2; kvp[3] = a3;
    if ((t & 7) == 0) ksum[bh * 32 + d] = ks;
}

// ---------------- y[token][h*32+m] = (phi(qg) . kv) / (phi(qg) . ksum + 1e-6) ----------------
__global__ __launch_bounds__(256) void y_kernel(const float* __restrict__ G1,
                                                const float* __restrict__ kv,
                                                const float* __restrict__ ksum,
                                                float* __restrict__ y) {
    int token = blockIdx.x, t = threadIdx.x;
    int h = t >> 5, m = t & 31;
    int b = token / Lx, bh = b * 8 + h;
    const float* qrow = G1 + (size_t)token * 2048 + 768 + h * 32;    // qg' at col 768
    const float* kvp  = kv + (size_t)bh * 1024;
    const float* ksp  = ksum + bh * 32;
    float num = 0, den = 0;
    #pragma unroll 8
    for (int d = 0; d < 32; ++d) {
        float qv = qrow[d];
        qv = qv > 0.0f ? qv + 1.0f : expf(qv);
        num += qv * kvp[d * 32 + m];
        den += qv * ksp[d];
    }
    y[(size_t)token * 256 + h * 32 + m] = num / (den + 1e-6f);
}

// ---------------- fusion weights + yc ----------------
__global__ __launch_bounds__(256) void fuse_kernel(const float* __restrict__ G1,
                                                   const float* __restrict__ wf2,
                                                   const float* __restrict__ bf2,
                                                   const float* __restrict__ h,
                                                   const float* __restrict__ hlocal,
                                                   const float* __restrict__ ygo,
                                                   const int* __restrict__ mask,
                                                   float* __restrict__ yc) {
    __shared__ float red[256];
    __shared__ float fw0s, fw1s;
    int token = blockIdx.x, t = threadIdx.x;
    float s = gelu_f(G1[(size_t)token * 2048 + 1536 + t]);   // f1 (biased) at col 1536
    float s0 = block_reduce_sum(s * wf2[t * 2 + 0], red);
    float s1 = block_reduce_sum(s * wf2[t * 2 + 1], red);
    if (t == 0) {
        float a = s0 + bf2[0], b = s1 + bf2[1];
        float mm = fmaxf(a, b);
        float ea = expf(a - mm), eb = expf(b - mm);
        float inv = 1.0f / (ea + eb);
        fw0s = ea * inv; fw1s = eb * inv;
    }
    __syncthreads();
    float mf = (float)mask[token];
    size_t i = (size_t)token * 256 + t;
    float hg = h[i] + ygo[i] * mf;                 // h_global
    yc[i] = fw0s * hlocal[i] + fw1s * hg;          // x2 == yc
}

// ---------------- out = (yc + ffn) * mf ----------------
__global__ void final_kernel(const float* __restrict__ yc, const float* __restrict__ ffn,
                             const int* __restrict__ mask, float* __restrict__ out) {
    size_t i = (size_t)blockIdx.x * blockDim.x + threadIdx.x;
    float mf = (float)mask[i >> 8];
    out[i] = (yc[i] + ffn[i]) * mf;
}

// =====================================================================
extern "C" void kernel_launch(void* const* d_in, const int* in_sizes, int n_in,
                              void* d_out, int out_size, void* d_ws, size_t ws_size,
                              hipStream_t stream) {
    const float* x      = (const float*)d_in[0];
    const int*   mask   = (const int*)  d_in[1];
    const int*   nbridx = (const int*)  d_in[2];
    const int*   nbrmsk = (const int*)  d_in[3];
    const int*   relpos = (const int*)  d_in[4];
    const float* g1     = (const float*)d_in[5];
    const float* b1     = (const float*)d_in[6];
    const float* wq     = (const float*)d_in[7];
    const float* wk     = (const float*)d_in[8];
    const float* wv     = (const float*)d_in[9];
    const float* remb   = (const float*)d_in[10];
    const float* we1    = (const float*)d_in[11];
    const float* be1    = (const float*)d_in[12];
    const float* we2    = (const float*)d_in[13];
    const float* be2    = (const float*)d_in[14];
    const float* wg1    = (const float*)d_in[15];
    const float* bg1    = (const float*)d_in[16];
    const float* wg2    = (const float*)d_in[17];
    const float* bg2    = (const float*)d_in[18];
    const float* wlo    = (const float*)d_in[19];
    const float* blo    = (const float*)d_in[20];
    const float* wqkv   = (const float*)d_in[21];
    const float* wgo    = (const float*)d_in[22];
    const float* wf1    = (const float*)d_in[23];
    const float* bf1    = (const float*)d_in[24];
    const float* wf2    = (const float*)d_in[25];
    const float* bf2    = (const float*)d_in[26];
    const float* g2     = (const float*)d_in[27];
    const float* b2     = (const float*)d_in[28];
    const float* wff1   = (const float*)d_in[29];
    const float* bff1   = (const float*)d_in[30];
    const float* wff2   = (const float*)d_in[31];
    const float* bff2   = (const float*)d_in[32];
    float* out = (float*)d_out;

    // -------- workspace carving --------
    char* base = (char*)d_ws;
    size_t off = 0;
    auto alloc = [&](size_t bytes) -> char* {
        char* p = base + off;
        off = (off + bytes + 255) & ~(size_t)255;
        return p;
    };
    const size_t MD = (size_t)Mx * Dx * sizeof(float);
    float* h      = (float*)alloc(MD);
    float* G1out  = (float*)alloc((size_t)Mx * 2048 * sizeof(float));
    float* agg0   = (float*)alloc(MD);
    float* agg    = (float*)alloc(MD);
    float* gbuf   = (float*)alloc(MD);
    float* hloc   = (float*)alloc(MD);
    float* ybuf   = (float*)alloc(MD);
    float* ygo    = (float*)alloc(MD);
    float* yc     = (float*)alloc(MD);
    float* hn     = (float*)alloc(MD);
    float* ubuf   = (float*)alloc((size_t)Mx * 1024 * sizeof(float));
    float* ffn    = (float*)alloc(MD);
    float* kvbuf  = (float*)alloc(32 * 32 * 32 * sizeof(float));
    float* ksum   = (float*)alloc(32 * 32 * sizeof(float));
    float* table  = (float*)alloc(65 * sizeof(float));
    float* bias2k = (float*)alloc(2048 * sizeof(float));
    u32* WtCat = (u32*)alloc((size_t)2048 * 128 * sizeof(u32));
    u32* Wtg1b = (u32*)alloc((size_t)256 * 128 * sizeof(u32));
    u32* Wtwlo = (u32*)alloc((size_t)256 * 128 * sizeof(u32));
    u32* Wtg2  = (u32*)alloc((size_t)256 * 128 * sizeof(u32));
    u32* Wtgo  = (u32*)alloc((size_t)256 * 128 * sizeof(u32));
    u32* Wtff1 = (u32*)alloc((size_t)1024 * 128 * sizeof(u32));
    u32* Wtff2 = (u32*)alloc((size_t)256 * 512 * sizeof(u32));
    (void)ws_size; (void)in_sizes; (void)n_in; (void)out_size;

    // -------- weight prep (bf16 pack, [n][K/2]) --------
    auto prep = [&](const float* src, int stride, int K, int N, u32* dst) {
        int tot = N * (K >> 1);
        prep_w<<<(tot + 255) / 256, 256, 0, stream>>>(src, stride, K, N, dst, K >> 1);
    };
    prep(wq,   256, 256, 256,  WtCat + (size_t)0    * 128);
    prep(wk,   256, 256, 256,  WtCat + (size_t)256  * 128);
    prep(wv,   256, 256, 256,  WtCat + (size_t)512  * 128);
    prep(wqkv, 768, 256, 768,  WtCat + (size_t)768  * 128);
    prep(wf1,  256, 256, 256,  WtCat + (size_t)1536 * 128);
    prep(wg1,  256, 256, 256,  WtCat + (size_t)1792 * 128);        // wg1 top half (rows 0..255)
    prep(wg1 + 256 * 256, 256, 256, 256, Wtg1b);                   // wg1 bottom half
    prep(wlo,  256, 256, 256,  Wtwlo);
    prep(wg2,  256, 256, 256,  Wtg2);
    prep(wgo,  256, 256, 256,  Wtgo);
    prep(wff1, 1024, 256, 1024, Wtff1);
    prep(wff2, 256, 1024, 256, Wtff2);
    biascat_kernel<<<8, 256, 0, stream>>>(bf1, bg1, bias2k);
    relbias_kernel<<<65, 256, 0, stream>>>(remb, we1, be1, we2, be2, table);

    // -------- h = LN(x) --------
    ln_kernel<<<Mx, 256, 0, stream>>>(x, g1, b1, h);

    // -------- big GEMM: [q|hk|hv|qkv|f1|g1h] = h @ Wcat (+biascat) --------
    gemm_bf16<<<dim3(2048 / 128, Mx / 128), 256, 0, stream>>>(h, 256, WtCat, 128, bias2k,
                                                              G1out, 2048, Mx, 2048, 256, 0);

    // -------- neighbor attention --------
    attn_kernel<<<Mx / 8, 256, 0, stream>>>(G1out, nbridx, nbrmsk, mask, relpos, table, agg0);

    // -------- agg = agg0 @ wlo + blo --------
    gemm_bf16<<<dim3(256 / 128, Mx / 128), 256, 0, stream>>>(agg0, 256, Wtwlo, 128, blo,
                                                             agg, 256, Mx, 256, 256, 0);
    // -------- t = gelu(h@wg1_top + bg1 + agg@wg1_bot)  (accum into G1out col 1792, fused gelu) --------
    gemm_bf16<<<dim3(256 / 128, Mx / 128), 256, 0, stream>>>(agg, 256, Wtg1b, 128, nullptr,
                                                             G1out + 1792, 2048, Mx, 256, 256, 1 | (1 << 1));
    // -------- g = sigmoid(t @ wg2 + bg2) --------
    gemm_bf16<<<dim3(256 / 128, Mx / 128), 256, 0, stream>>>(G1out + 1792, 2048, Wtg2, 128, bg2,
                                                             gbuf, 256, Mx, 256, 256, (2 << 1));
    // -------- h_local --------
    hlocal_kernel<<<Mx, 256, 0, stream>>>(h, gbuf, agg, mask, hloc);

    // -------- linear attention --------
    kv_kernel<<<32, 256, 0, stream>>>(G1out, mask, kvbuf, ksum);
    y_kernel<<<Mx, 256, 0, stream>>>(G1out, kvbuf, ksum, ybuf);
    gemm_bf16<<<dim3(256 / 128, Mx / 128), 256, 0, stream>>>(ybuf, 256, Wtgo, 128, nullptr,
                                                             ygo, 256, Mx, 256, 256, 0);

    // -------- fusion: yc (== x2) --------
    fuse_kernel<<<Mx, 256, 0, stream>>>(G1out, wf2, bf2, h, hloc, ygo, mask, yc);

    // -------- FFN --------
    ln_kernel<<<Mx, 256, 0, stream>>>(yc, g2, b2, hn);
    gemm_bf16<<<dim3(1024 / 128, Mx / 128), 256, 0, stream>>>(hn, 256, Wtff1, 128, bff1,
                                                              ubuf, 1024, Mx, 1024, 256, (1 << 1));
    gemm_bf16<<<dim3(256 / 128, Mx / 128), 256, 0, stream>>>(ubuf, 1024, Wtff2, 512, bff2,
                                                             ffn, 256, Mx, 256, 1024, 0);

    // -------- out = (yc + ffn) * mf --------
    final_kernel<<<Mx, 256, 0, stream>>>(yc, ffn, mask, out);
}
